// TransformerEncoderLayerWithBias_22978075034135
// MI455X (gfx1250) — matrix operand order repaired and checked by reading, not compiled, not run on hardware
//
#include <hip/hip_runtime.h>
#include <math.h>

typedef _Float16 half_t;
typedef __attribute__((ext_vector_type(16))) _Float16 v16h;
typedef __attribute__((ext_vector_type(8)))  _Float16 v8h;
typedef __attribute__((ext_vector_type(8)))  float    v8f;

union F16x16 { v16h v; v8h h[2]; };

__device__ __forceinline__ v8h load8h(const half_t* p) {
    v8h r;
    __builtin_memcpy(&r, __builtin_assume_aligned(p, 16), 16);
    return r;
}
__device__ __forceinline__ void store8h(half_t* p, v8h v) {
    __builtin_memcpy(__builtin_assume_aligned(p, 16), &v, 16);
}

// ---------------------------------------------------------------------------
// CDNA5 async global->LDS copy (ASYNCcnt-tracked), with safe fallback.
// ROCm 7.2 clang-22 types the builtin as:
//   void __builtin_amdgcn_global_load_async_to_lds_b128(
//       v4i AS(1)* gsrc, v4i AS(3)* ldst, int offset, int cpol)
// ---------------------------------------------------------------------------
#if defined(__gfx1250__) && __has_builtin(__builtin_amdgcn_global_load_async_to_lds_b128)
#define HAVE_ASYNC_LDS 1
#else
#define HAVE_ASYNC_LDS 0
#endif

#if HAVE_ASYNC_LDS
typedef int v4i_t __attribute__((__vector_size__(16)));
typedef __attribute__((address_space(1))) v4i_t g_v4i;
typedef __attribute__((address_space(3))) v4i_t l_v4i;
__device__ __forceinline__ void async_copy16(half_t* l, const half_t* g) {
    __builtin_amdgcn_global_load_async_to_lds_b128((g_v4i*)g, (l_v4i*)l, 0, 0);
}
__device__ __forceinline__ void wait_async0() {
#if __has_builtin(__builtin_amdgcn_s_wait_asynccnt)
    __builtin_amdgcn_s_wait_asynccnt(0);
#else
    asm volatile("s_wait_asynccnt 0x0" ::: "memory");
#endif
}
#else
__device__ __forceinline__ void async_copy16(half_t* l, const half_t* g) {
    store8h(l, load8h(g));
}
__device__ __forceinline__ void wait_async0() {}
#endif

// ---------------------------------------------------------------------------
// f32 -> f16 conversion (weights), grid-stride
// ---------------------------------------------------------------------------
__global__ void cvt_f32_f16_kernel(const float* __restrict__ src,
                                   half_t* __restrict__ dst, int n) {
    int i = blockIdx.x * blockDim.x + threadIdx.x;
    int stride = gridDim.x * blockDim.x;
    for (; i < n; i += stride) dst[i] = (half_t)src[i];
}

// ---------------------------------------------------------------------------
// LayerNorm over last dim (1024), one block per row, writes f16
// ---------------------------------------------------------------------------
__global__ __launch_bounds__(256)
void layernorm_kernel(const float* __restrict__ x, const float* __restrict__ w,
                      const float* __restrict__ b, half_t* __restrict__ out) {
    const int row = blockIdx.x;
    const int tid = threadIdx.x;
    const int lane = tid & 31;
    const int wid = tid >> 5;
    const float* xr = x + (size_t)row * 1024;

    float v[4], s = 0.f, s2 = 0.f;
#pragma unroll
    for (int i = 0; i < 4; ++i) {
        v[i] = xr[tid + i * 256];
        s += v[i];
        s2 += v[i] * v[i];
    }
#pragma unroll
    for (int m = 16; m >= 1; m >>= 1) {
        s  += __shfl_xor(s,  m, 32);
        s2 += __shfl_xor(s2, m, 32);
    }
    __shared__ float red[16];
    if (lane == 0) { red[wid] = s; red[8 + wid] = s2; }
    __syncthreads();
    if (tid == 0) {
        float a = 0.f, c = 0.f;
#pragma unroll
        for (int i = 0; i < 8; ++i) { a += red[i]; c += red[8 + i]; }
        red[0] = a; red[8] = c;
    }
    __syncthreads();
    const float mu = red[0] * (1.0f / 1024.0f);
    const float var = red[8] * (1.0f / 1024.0f) - mu * mu;
    const float rs = rsqrtf(var + 1e-5f);
#pragma unroll
    for (int i = 0; i < 4; ++i) {
        int c = tid + i * 256;
        out[(size_t)row * 1024 + c] = (half_t)((v[i] - mu) * rs * w[c] + b[c]);
    }
}

// ---------------------------------------------------------------------------
// WMMA GEMM: C[M,N] = A[M,K] * B[N,K]^T  (+ epilogue)
//   Block tile 128x128, BK=32, double-buffered LDS with async global->LDS.
//   8 waves, wave tile 32x64 (2x4 WMMA 16x16 tiles), 8 WMMAs per wave per step.
//   EPI 0: store f16 (acc + bias)
//   EPI 1: store f16 gelu(acc + bias)
//   EPI 2: store f32 (acc + bias + res)
// ---------------------------------------------------------------------------
#define LDT 40  // padded LDS stride in halves for 32-wide k tiles

template <int EPI>
__global__ __launch_bounds__(256)
void gemm_wmma_kernel(const half_t* __restrict__ A, const half_t* __restrict__ B,
                      const float* __restrict__ bias, const float* res,
                      void* out, int M, int N, int K) {
    __shared__ half_t Ash[2][128 * LDT];
    __shared__ half_t Bsh[2][128 * LDT];

    const int tid = threadIdx.x;
    const int lane = tid & 31;
    const int wid = tid >> 5;
    const int l15 = lane & 15;
    const int hi = lane >> 4;           // 0 or 1
    const int wm = (wid >> 1) * 32;     // wave row within block tile
    const int wn = (wid & 1) * 64;      // wave col within block tile
    const int bm = blockIdx.x * 128;
    const int bn = blockIdx.y * 128;

    const v8f zacc = {0.f, 0.f, 0.f, 0.f, 0.f, 0.f, 0.f, 0.f};
    v8f acc[2][4];
#pragma unroll
    for (int mi = 0; mi < 2; ++mi)
#pragma unroll
        for (int ni = 0; ni < 4; ++ni) acc[mi][ni] = zacc;

    // stage one 128x32 A tile + 128x32 B tile into buffer `buf`
    auto stage = [&](int buf, int k0) {
#pragma unroll
        for (int c = tid; c < 512; c += 256) {
            int row = c >> 2, kc = (c & 3) * 8;
            async_copy16(&Ash[buf][row * LDT + kc],
                         &A[(size_t)(bm + row) * K + k0 + kc]);
        }
#pragma unroll
        for (int c = tid; c < 512; c += 256) {
            int row = c >> 2, kc = (c & 3) * 8;
            async_copy16(&Bsh[buf][row * LDT + kc],
                         &B[(size_t)(bn + row) * K + k0 + kc]);
        }
    };

    stage(0, 0);
    wait_async0();
    __syncthreads();

    int buf = 0;
    for (int k0 = 0; k0 < K; k0 += 32) {
        const int nxt = buf ^ 1;
        if (k0 + 32 < K) stage(nxt, k0 + 32);  // stream next tile during compute

        F16x16 af[2], bf[4];
#pragma unroll
        for (int mi = 0; mi < 2; ++mi) {
            const half_t* p = &Ash[buf][(wm + mi * 16 + l15) * LDT + hi * 8];
            af[mi].h[0] = load8h(p);
            af[mi].h[1] = load8h(p + 16);
        }
#pragma unroll
        for (int ni = 0; ni < 4; ++ni) {
            const half_t* p = &Bsh[buf][(wn + ni * 16 + l15) * LDT + hi * 8];
            bf[ni].h[0] = load8h(p);
            bf[ni].h[1] = load8h(p + 16);
        }
#pragma unroll
        for (int mi = 0; mi < 2; ++mi)
#pragma unroll
            for (int ni = 0; ni < 4; ++ni)
                acc[mi][ni] = __builtin_amdgcn_wmma_f32_16x16x32_f16(
                    false, af[mi].v, false, bf[ni].v, (short)0, acc[mi][ni],
                    false, false);

        wait_async0();
        __syncthreads();
        buf = nxt;
    }

    // epilogue
#pragma unroll
    for (int mi = 0; mi < 2; ++mi) {
#pragma unroll
        for (int ni = 0; ni < 4; ++ni) {
            const int col = bn + wn + ni * 16 + l15;
            const float bv = bias[col];
#pragma unroll
            for (int r = 0; r < 8; ++r) {
                const int row = bm + wm + mi * 16 + r + hi * 8;
                const size_t idx = (size_t)row * N + col;
                float t = acc[mi][ni][r] + bv;
                if (EPI == 0) {
                    ((half_t*)out)[idx] = (half_t)t;
                } else if (EPI == 1) {
                    t = 0.5f * t * (1.0f + erff(t * 0.70710678118654752f));
                    ((half_t*)out)[idx] = (half_t)t;
                } else {
                    ((float*)out)[idx] = t + res[idx];
                }
            }
        }
    }
}

// ---------------------------------------------------------------------------
// Flash attention with rel-pos bias (bias row staged in LDS).
//   qkv: f16 [B*S, 3072] rows; Q at col 0, K at 1024, V at 2048 (h*64 + e).
//   One block = (b, h, 128 q rows); 8 waves x 16 q rows each.
// ---------------------------------------------------------------------------
__global__ __launch_bounds__(256)
void attn_kernel(const half_t* __restrict__ qkv,
                 const float* __restrict__ rel_bias,
                 half_t* __restrict__ attn_out) {
    const int bid = blockIdx.x;
    const int qt = bid & 3;
    const int h  = (bid >> 2) & 15;
    const int b  = bid >> 6;
    const int q0 = qt * 128;

    const int tid = threadIdx.x;
    const int lane = tid & 31;
    const int wid = tid >> 5;
    const int l15 = lane & 15;
    const int hi = lane >> 4;

    __shared__ half_t Vsh[64 * LDT];       // V^T chunk: [e][kv], 32 kv
    __shared__ half_t Psh[8][16 * LDT];    // per-wave P tile 16x32
    __shared__ float  BiasSh[1024];        // rel_bias row for this head

    for (int i = tid; i < 1023; i += 256) BiasSh[i] = rel_bias[h * 1023 + i];

    const v8f zacc = {0.f, 0.f, 0.f, 0.f, 0.f, 0.f, 0.f, 0.f};

    // Q fragments for this wave's 16 rows (A-fragment layout, e-chunks 0/32)
    const half_t* qbase =
        qkv + ((size_t)(b * 512 + q0 + wid * 16 + l15)) * 3072 + h * 64;
    F16x16 qf0, qf1;
    qf0.h[0] = load8h(qbase + hi * 8);
    qf0.h[1] = load8h(qbase + 16 + hi * 8);
    qf1.h[0] = load8h(qbase + 32 + hi * 8);
    qf1.h[1] = load8h(qbase + 48 + hi * 8);

    float m_r[8], l_r[8];
    v8f O[4];
#pragma unroll
    for (int r = 0; r < 8; ++r) { m_r[r] = -3.0e38f; l_r[r] = 0.f; }
#pragma unroll
    for (int ni = 0; ni < 4; ++ni) O[ni] = zacc;

    __syncthreads();  // BiasSh ready

    for (int kv0 = 0; kv0 < 512; kv0 += 32) {
        // stage V chunk transposed: Vsh[e][kk]
        {
            int kk = tid >> 3;
            int eg = (tid & 7) * 8;
            v8h vv = load8h(
                &qkv[((size_t)(b * 512 + kv0 + kk)) * 3072 + 2048 + h * 64 + eg]);
#pragma unroll
            for (int i = 0; i < 8; ++i) Vsh[(eg + i) * LDT + kk] = vv[i];
        }
        __syncthreads();

        // scores: two 16x16 tiles covering kv0..kv0+31
        v8f s[2];
#pragma unroll
        for (int t = 0; t < 2; ++t) {
            const half_t* kp =
                &qkv[((size_t)(b * 512 + kv0 + t * 16 + l15)) * 3072 + 1024 +
                     h * 64];
            F16x16 b0, b1;
            b0.h[0] = load8h(kp + hi * 8);
            b0.h[1] = load8h(kp + 16 + hi * 8);
            b1.h[0] = load8h(kp + 32 + hi * 8);
            b1.h[1] = load8h(kp + 48 + hi * 8);
            s[t] = __builtin_amdgcn_wmma_f32_16x16x32_f16(
                false, qf0.v, false, b0.v, (short)0, zacc, false, false);
            s[t] = __builtin_amdgcn_wmma_f32_16x16x32_f16(
                false, qf1.v, false, b1.v, (short)0, s[t], false, false);
        }

        // streaming softmax update
        float p0[8], p1[8], sc[8];
#pragma unroll
        for (int r = 0; r < 8; ++r) {
            const int q = q0 + wid * 16 + r + hi * 8;
            int r0 = q - (kv0 + l15) + 511;
            int r1 = q - (kv0 + 16 + l15) + 511;
            r0 = r0 < 0 ? 0 : (r0 > 1022 ? 1022 : r0);
            r1 = r1 < 0 ? 0 : (r1 > 1022 ? 1022 : r1);
            float v0 = s[0][r] * 0.125f + BiasSh[r0];
            float v1 = s[1][r] * 0.125f + BiasSh[r1];
            float cm = fmaxf(v0, v1);
            cm = fmaxf(cm, __shfl_xor(cm, 1, 32));
            cm = fmaxf(cm, __shfl_xor(cm, 2, 32));
            cm = fmaxf(cm, __shfl_xor(cm, 4, 32));
            cm = fmaxf(cm, __shfl_xor(cm, 8, 32));
            const float newm = fmaxf(m_r[r], cm);
            sc[r] = __expf(m_r[r] - newm);
            p0[r] = __expf(v0 - newm);
            p1[r] = __expf(v1 - newm);
            float rs = p0[r] + p1[r];
            rs += __shfl_xor(rs, 1, 32);
            rs += __shfl_xor(rs, 2, 32);
            rs += __shfl_xor(rs, 4, 32);
            rs += __shfl_xor(rs, 8, 32);
            l_r[r] = l_r[r] * sc[r] + rs;
            m_r[r] = newm;
        }
#pragma unroll
        for (int ni = 0; ni < 4; ++ni)
#pragma unroll
            for (int r = 0; r < 8; ++r) O[ni][r] *= sc[r];

        // P: C-fragment layout -> LDS -> A-fragment layout (f16)
#pragma unroll
        for (int r = 0; r < 8; ++r) {
            const int prow = r + hi * 8;
            Psh[wid][prow * LDT + l15] = (half_t)p0[r];
            Psh[wid][prow * LDT + 16 + l15] = (half_t)p1[r];
        }
        F16x16 pf;
        pf.h[0] = load8h(&Psh[wid][l15 * LDT + hi * 8]);
        pf.h[1] = load8h(&Psh[wid][l15 * LDT + 16 + hi * 8]);

#pragma unroll
        for (int ni = 0; ni < 4; ++ni) {
            F16x16 vf;
            const half_t* vp = &Vsh[(ni * 16 + l15) * LDT + hi * 8];
            vf.h[0] = load8h(vp);
            vf.h[1] = load8h(vp + 16);
            O[ni] = __builtin_amdgcn_wmma_f32_16x16x32_f16(
                false, pf.v, false, vf.v, (short)0, O[ni], false, false);
        }
        __syncthreads();
    }

    // normalize + store attention output, layout [B*S, 1024] f16
#pragma unroll
    for (int r = 0; r < 8; ++r) {
        const float inv = 1.0f / l_r[r];
        const size_t row = (size_t)(b * 512 + q0 + wid * 16 + r + hi * 8);
#pragma unroll
        for (int ni = 0; ni < 4; ++ni)
            attn_out[row * 1024 + h * 64 + ni * 16 + l15] =
                (half_t)(O[ni][r] * inv);
    }
}

// ---------------------------------------------------------------------------
// Launch
// ---------------------------------------------------------------------------
extern "C" void kernel_launch(void* const* d_in, const int* in_sizes, int n_in,
                              void* d_out, int out_size, void* d_ws, size_t ws_size,
                              hipStream_t stream) {
    (void)in_sizes; (void)n_in; (void)out_size; (void)ws_size;
    const int D = 1024, DFF = 4096, M = 16 * 512;

    const float* x        = (const float*)d_in[0];
    const float* in_w     = (const float*)d_in[1];
    const float* in_b     = (const float*)d_in[2];
    const float* out_w    = (const float*)d_in[3];
    const float* out_b    = (const float*)d_in[4];
    const float* rel_bias = (const float*)d_in[5];
    const float* w1       = (const float*)d_in[6];
    const float* b1       = (const float*)d_in[7];
    const float* w2       = (const float*)d_in[8];
    const float* b2       = (const float*)d_in[9];
    const float* ln1_w    = (const float*)d_in[10];
    const float* ln1_b    = (const float*)d_in[11];
    const float* ln2_w    = (const float*)d_in[12];
    const float* ln2_b    = (const float*)d_in[13];
    float* outf = (float*)d_out;

    // workspace layout (halves)
    half_t* ws = (half_t*)d_ws;
    half_t* w_in_h  = ws;                           // 3*D*D
    half_t* w_out_h = w_in_h + (size_t)3 * D * D;   // D*D
    half_t* w1_h    = w_out_h + (size_t)D * D;      // DFF*D
    half_t* w2_h    = w1_h + (size_t)DFF * D;       // D*DFF
    half_t* xh      = w2_h + (size_t)D * DFF;       // M*D (ln1 out, reused for ln2)
    half_t* qkv_h   = xh + (size_t)M * D;           // max(M*3D, M*DFF): qkv / ffn hidden
    half_t* attn_h  = qkv_h + (size_t)M * DFF;      // M*D

    // 1) weight conversion f32 -> f16
    cvt_f32_f16_kernel<<<2048, 256, 0, stream>>>(in_w,  w_in_h,  3 * D * D);
    cvt_f32_f16_kernel<<<2048, 256, 0, stream>>>(out_w, w_out_h, D * D);
    cvt_f32_f16_kernel<<<2048, 256, 0, stream>>>(w1,    w1_h,    DFF * D);
    cvt_f32_f16_kernel<<<2048, 256, 0, stream>>>(w2,    w2_h,    D * DFF);

    // 2) LN1
    layernorm_kernel<<<M, 256, 0, stream>>>(x, ln1_w, ln1_b, xh);

    // 3) QKV projection: [M,3072] = xh[M,1024] * in_w[3072,1024]^T
    gemm_wmma_kernel<0><<<dim3(M / 128, 3072 / 128), 256, 0, stream>>>(
        xh, w_in_h, in_b, nullptr, qkv_h, M, 3 * D, D);

    // 4) attention
    attn_kernel<<<1024, 256, 0, stream>>>(qkv_h, rel_bias, attn_h);

    // 5) out projection + residual -> d_out (f32)
    gemm_wmma_kernel<2><<<dim3(M / 128, D / 128), 256, 0, stream>>>(
        attn_h, w_out_h, out_b, x, outf, M, D, D);

    // 6) LN2
    layernorm_kernel<<<M, 256, 0, stream>>>(outf, ln2_w, ln2_b, xh);

    // 7) FFN up + exact GELU -> f16 hidden (reuse qkv region)
    gemm_wmma_kernel<1><<<dim3(M / 128, DFF / 128), 256, 0, stream>>>(
        xh, w1_h, b1, nullptr, qkv_h, M, DFF, D);

    // 8) FFN down + residual -> d_out
    gemm_wmma_kernel<2><<<dim3(M / 128, D / 128), 256, 0, stream>>>(
        qkv_h, w2_h, b2, outf, outf, M, D, DFF);
}
